// FeatUpdate_901943132400
// MI455X (gfx1250) — matrix operand order repaired and checked by reading, not compiled, not run on hardware
//
#include <hip/hip_runtime.h>
#include <hip/hip_bf16.h>

// ---------------- problem constants ----------------
#define NNODE 10000
#define NEDGE 160000
#define IND   256
#define HIDD  512
#define EAD   16
#define K1    528          // 2*IND + EAD
#define K1P   544          // K1 padded to multiple of 32 (LDS pitch / w1T row pitch)
#define M1P   520          // LDS pitch (halfs) for 512-wide activation tiles (16B-aligned rows)
#define K3    768          // IND + HIDD (already multiple of 32)
#define K3P   776          // LDS pitch for node input tile
#define MT    32           // rows (edges/nodes) per workgroup: 2 WMMA M-tiles

typedef __attribute__((ext_vector_type(16))) _Float16 v16h;
typedef __attribute__((ext_vector_type(8)))  _Float16 v8h;
typedef __attribute__((ext_vector_type(8)))  float    v8f;

__device__ __forceinline__ float silu_f(float x) {
    return x / (1.0f + __expf(-x));
}

// convert 16 consecutive f32 (as 4 float4) -> 16 f16, store as two 16B LDS stores
__device__ __forceinline__ void store16_h(_Float16* dst, const float4* src, float scale) {
    float4 f0 = src[0], f1 = src[1], f2 = src[2], f3 = src[3];
    v8h a, b;
    a[0] = (_Float16)(f0.x * scale); a[1] = (_Float16)(f0.y * scale);
    a[2] = (_Float16)(f0.z * scale); a[3] = (_Float16)(f0.w * scale);
    a[4] = (_Float16)(f1.x * scale); a[5] = (_Float16)(f1.y * scale);
    a[6] = (_Float16)(f1.z * scale); a[7] = (_Float16)(f1.w * scale);
    b[0] = (_Float16)(f2.x * scale); b[1] = (_Float16)(f2.y * scale);
    b[2] = (_Float16)(f2.z * scale); b[3] = (_Float16)(f2.w * scale);
    b[4] = (_Float16)(f3.x * scale); b[5] = (_Float16)(f3.y * scale);
    b[6] = (_Float16)(f3.z * scale); b[7] = (_Float16)(f3.w * scale);
    *(v8h*)dst       = a;
    *(v8h*)(dst + 8) = b;
}

__device__ __forceinline__ void store16_zero(_Float16* dst) {
    v8h z = (v8h)(_Float16)0.0f;
    *(v8h*)dst       = z;
    *(v8h*)(dst + 8) = z;
}

// A-fragment (16x32 f16): elems 0..15 = k(hl*16 + 0..15), matching B's contiguous order
__device__ __forceinline__ v16h load_afrag(const _Float16* rowk, int hl) {
    union { v16h v; v8h h2[2]; } u;
    u.h2[0] = *(const v8h*)(rowk + hl * 16);
    u.h2[1] = *(const v8h*)(rowk + hl * 16 + 8);
    return u.v;
}

#define WMMA_F16(A, B, C) \
    __builtin_amdgcn_wmma_f32_16x16x32_f16(false, (A), false, (B), (short)0, (C), false, false)

// ---------------- prep kernels ----------------
__global__ void zero_f32_kernel(float* p, int n) {
    int i = blockIdx.x * 256 + threadIdx.x;
    if (i < n) p[i] = 0.0f;
}

// wT[n][k] (f16, row pitch Kpad) = w[k][n] (f32), zero-padded for k >= K
__global__ void cvt_transpose_kernel(const float* __restrict__ w, _Float16* __restrict__ wT,
                                     int K, int Nout, int Kpad) {
    int idx = blockIdx.x * 256 + threadIdx.x;
    if (idx >= Nout * Kpad) return;
    int n = idx / Kpad;
    int k = idx - n * Kpad;
    wT[idx] = (k < K) ? (_Float16)w[(size_t)k * Nout + n] : (_Float16)0.0f;
}

// ---------------- edge message kernel ----------------
// grid = NEDGE/32 blocks, 256 threads (8 waves). Wave w owns HID columns [64w,64w+64),
// and processes 2 M-tiles (32 edges) so each B fragment feeds 2 WMMAs.
__global__ __launch_bounds__(256) void edge_msg_kernel(
    const float* __restrict__ h, const int* __restrict__ ei, const float* __restrict__ ea,
    const _Float16* __restrict__ w1T, const float* __restrict__ b1,
    const _Float16* __restrict__ w2T, const float* __restrict__ b2,
    const float* __restrict__ wa, const float* __restrict__ ba,
    float* __restrict__ agg)
{
    __shared__ _Float16 sx[MT * K1P];    // [edge][k] f16 input tile (~34.8 KB)
    __shared__ _Float16 sm1[MT * M1P];   // layer-1 activations (~33.3 KB)
    __shared__ float    satt[MT];        // attention logit accumulator
    __shared__ int      srow[MT];        // destination node per edge

    const int tid = threadIdx.x;
    const int e0  = blockIdx.x * MT;

    // ---- gather: x = [h[row] | h[col] | ea | 0-pad] as f16 into LDS ----
    {
        const int e = tid >> 3;          // edge within tile (0..31)
        const int l = tid & 7;           // 8 threads per edge
        const int r = ei[e0 + e];
        const int c = ei[NEDGE + e0 + e];
        if (l == 0) { srow[e] = r; satt[e] = ba[0]; }
        _Float16* dst = sx + e * K1P;
        const float* hr = h + (size_t)r * IND + l * 32;
        const float* hc = h + (size_t)c * IND + l * 32;
        store16_h(dst + l * 32,            (const float4*)hr,        1.0f);
        store16_h(dst + l * 32 + 16,       (const float4*)(hr + 16), 1.0f);
        store16_h(dst + IND + l * 32,      (const float4*)hc,        1.0f);
        store16_h(dst + IND + l * 32 + 16, (const float4*)(hc + 16), 1.0f);
        if (l == 0) {
            store16_h(dst + 2 * IND, (const float4*)(ea + (size_t)(e0 + e) * EAD), 1.0f);
            store16_zero(dst + K1);  // pad 528..543
        }
    }
    __syncthreads();

    const int lane = tid & 31;
    const int wv   = tid >> 5;          // wave id 0..7
    const int m    = lane & 15;         // A row select / C column lane
    const int hl   = lane >> 4;         // lane half
    const int nb0  = wv * 64;

    // ---- layer 1: m1 = silu(x @ w1 + b1) ----
    {
        v8f acc[2][4] = {};
        for (int kb = 0; kb < K1P; kb += 32) {
            v16h a0 = load_afrag(sx + m * K1P + kb, hl);
            v16h a1 = load_afrag(sx + (16 + m) * K1P + kb, hl);
            #pragma unroll
            for (int t = 0; t < 4; ++t) {
                v16h b = *(const v16h*)(w1T + (size_t)(nb0 + t * 16 + m) * K1P + kb + hl * 16);
                acc[0][t] = WMMA_F16(a0, b, acc[0][t]);
                acc[1][t] = WMMA_F16(a1, b, acc[1][t]);
            }
        }
        #pragma unroll
        for (int mi = 0; mi < 2; ++mi) {
            #pragma unroll
            for (int t = 0; t < 4; ++t) {
                int n = nb0 + t * 16 + m;
                float bias = b1[n];
                #pragma unroll
                for (int v = 0; v < 8; ++v) {
                    int row = mi * 16 + v + 8 * hl;      // C layout: M = v + 8*half
                    sm1[row * M1P + n] = (_Float16)silu_f(acc[mi][t][v] + bias);
                }
            }
        }
    }
    __syncthreads();

    // ---- layer 2: m2 = silu(m1 @ w2 + b2), attention, scatter ----
    v8f acc2[2][4] = {};
    for (int kb = 0; kb < HIDD; kb += 32) {
        v16h a0 = load_afrag(sm1 + m * M1P + kb, hl);
        v16h a1 = load_afrag(sm1 + (16 + m) * M1P + kb, hl);
        #pragma unroll
        for (int t = 0; t < 4; ++t) {
            v16h b = *(const v16h*)(w2T + (size_t)(nb0 + t * 16 + m) * HIDD + kb + hl * 16);
            acc2[0][t] = WMMA_F16(a0, b, acc2[0][t]);
            acc2[1][t] = WMMA_F16(a1, b, acc2[1][t]);
        }
    }

    // epilogue: silu in-place + attention-logit partials (reduced via ds_add_f32)
    {
        float part[2][8];
        #pragma unroll
        for (int mi = 0; mi < 2; ++mi)
            #pragma unroll
            for (int v = 0; v < 8; ++v) part[mi][v] = 0.0f;
        #pragma unroll
        for (int t = 0; t < 4; ++t) {
            int n = nb0 + t * 16 + m;
            float bias = b2[n];
            float wan  = wa[n];
            #pragma unroll
            for (int mi = 0; mi < 2; ++mi) {
                #pragma unroll
                for (int v = 0; v < 8; ++v) {
                    float x = silu_f(acc2[mi][t][v] + bias);
                    acc2[mi][t][v] = x;              // keep gated-input in regs
                    part[mi][v] += x * wan;
                }
            }
        }
        #pragma unroll
        for (int mi = 0; mi < 2; ++mi)
            #pragma unroll
            for (int v = 0; v < 8; ++v)
                atomicAdd(&satt[mi * 16 + v + 8 * hl], part[mi][v]);
    }
    __syncthreads();

    // ---- gate and scatter-add into agg (L2-resident, relaxed fp32 atomics) ----
    #pragma unroll
    for (int mi = 0; mi < 2; ++mi) {
        #pragma unroll
        for (int v = 0; v < 8; ++v) {
            int row = mi * 16 + v + 8 * hl;
            float att = 1.0f / (1.0f + __expf(-satt[row]));
            int dstn  = srow[row];
            #pragma unroll
            for (int t = 0; t < 4; ++t) {
                int n = nb0 + t * 16 + m;
                __hip_atomic_fetch_add(&agg[(size_t)dstn * HIDD + n], acc2[mi][t][v] * att,
                                       __ATOMIC_RELAXED, __HIP_MEMORY_SCOPE_AGENT);
            }
        }
    }
}

// ---------------- node update kernel ----------------
// grid = ceil(NNODE/32) blocks, 256 threads. 32 nodes per WG (tail-guarded).
__global__ __launch_bounds__(256) void node_update_kernel(
    const float* __restrict__ h, const float* __restrict__ agg,
    const _Float16* __restrict__ w3T, const float* __restrict__ b3,
    const _Float16* __restrict__ w4T, const float* __restrict__ b4,
    float* __restrict__ out)
{
    __shared__ _Float16 su[MT * K3P];    // [node][k]: [h | agg/100] (~49.7 KB)
    __shared__ _Float16 sm3[MT * M1P];   // layer-3 activations (~33.3 KB)

    const int tid = threadIdx.x;
    const int n0  = blockIdx.x * MT;

    // ---- gather concat(h, agg/NORM) as f16 (zero-fill past NNODE) ----
    {
        const int nd = tid >> 3;         // node within tile (0..31)
        const int l  = tid & 7;          // 8 threads per node
        const size_t node = n0 + nd;
        _Float16* dst = su + nd * K3P;
        if (node < NNODE) {
            const float* hp = h + node * IND + l * 32;
            store16_h(dst + l * 32,      (const float4*)hp,        1.0f);
            store16_h(dst + l * 32 + 16, (const float4*)(hp + 16), 1.0f);
            const float* ag = agg + node * HIDD + l * 64;
            store16_h(dst + IND + l * 64,      (const float4*)ag,        0.01f); // 1/NORM
            store16_h(dst + IND + l * 64 + 16, (const float4*)(ag + 16), 0.01f);
            store16_h(dst + IND + l * 64 + 32, (const float4*)(ag + 32), 0.01f);
            store16_h(dst + IND + l * 64 + 48, (const float4*)(ag + 48), 0.01f);
        } else {
            store16_zero(dst + l * 32);
            store16_zero(dst + IND + l * 64);
            store16_zero(dst + IND + l * 64 + 32);
        }
    }
    __syncthreads();

    const int lane = tid & 31;
    const int wv   = tid >> 5;
    const int m    = lane & 15;
    const int hl   = lane >> 4;

    // ---- layer 3: u = silu([h|agg] @ w3 + b3) ----
    {
        const int nb0 = wv * 64;
        v8f acc[2][4] = {};
        for (int kb = 0; kb < K3; kb += 32) {
            v16h a0 = load_afrag(su + m * K3P + kb, hl);
            v16h a1 = load_afrag(su + (16 + m) * K3P + kb, hl);
            #pragma unroll
            for (int t = 0; t < 4; ++t) {
                v16h b = *(const v16h*)(w3T + (size_t)(nb0 + t * 16 + m) * K3 + kb + hl * 16);
                acc[0][t] = WMMA_F16(a0, b, acc[0][t]);
                acc[1][t] = WMMA_F16(a1, b, acc[1][t]);
            }
        }
        #pragma unroll
        for (int mi = 0; mi < 2; ++mi) {
            #pragma unroll
            for (int t = 0; t < 4; ++t) {
                int n = nb0 + t * 16 + m;
                float bias = b3[n];
                #pragma unroll
                for (int v = 0; v < 8; ++v)
                    sm3[(mi * 16 + v + 8 * hl) * M1P + n] = (_Float16)silu_f(acc[mi][t][v] + bias);
            }
        }
    }
    __syncthreads();

    // ---- layer 4 + residual: out = h + u @ w4 + b4 ----
    {
        const int nb0 = wv * 32;
        v8f acc[2][2] = {};
        for (int kb = 0; kb < HIDD; kb += 32) {
            v16h a0 = load_afrag(sm3 + m * M1P + kb, hl);
            v16h a1 = load_afrag(sm3 + (16 + m) * M1P + kb, hl);
            #pragma unroll
            for (int t = 0; t < 2; ++t) {
                v16h b = *(const v16h*)(w4T + (size_t)(nb0 + t * 16 + m) * HIDD + kb + hl * 16);
                acc[0][t] = WMMA_F16(a0, b, acc[0][t]);
                acc[1][t] = WMMA_F16(a1, b, acc[1][t]);
            }
        }
        #pragma unroll
        for (int mi = 0; mi < 2; ++mi) {
            #pragma unroll
            for (int t = 0; t < 2; ++t) {
                int n = nb0 + t * 16 + m;
                float bias = b4[n];
                #pragma unroll
                for (int v = 0; v < 8; ++v) {
                    size_t node = n0 + mi * 16 + v + 8 * hl;
                    if (node < NNODE)
                        out[node * IND + n] = h[node * IND + n] + acc[mi][t][v] + bias;
                }
            }
        }
    }
}

// ---------------- launch ----------------
extern "C" void kernel_launch(void* const* d_in, const int* in_sizes, int n_in,
                              void* d_out, int out_size, void* d_ws, size_t ws_size,
                              hipStream_t stream) {
    const float* h  = (const float*)d_in[0];
    const int*   ei = (const int*)d_in[1];
    const float* ea = (const float*)d_in[2];
    const float* w1 = (const float*)d_in[3];
    const float* b1 = (const float*)d_in[4];
    const float* w2 = (const float*)d_in[5];
    const float* b2 = (const float*)d_in[6];
    const float* wa = (const float*)d_in[7];
    const float* ba = (const float*)d_in[8];
    const float* w3 = (const float*)d_in[9];
    const float* b3 = (const float*)d_in[10];
    const float* w4 = (const float*)d_in[11];
    const float* b4 = (const float*)d_in[12];
    float* out = (float*)d_out;

    // workspace carve-up (all 256B-aligned)
    char* ws = (char*)d_ws;
    float*    agg = (float*)ws;                         size_t off = (size_t)NNODE * HIDD * 4;
    _Float16* w1T = (_Float16*)(ws + off);              off += (size_t)HIDD * K1P * 2;
    _Float16* w2T = (_Float16*)(ws + off);              off += (size_t)HIDD * HIDD * 2;
    _Float16* w3T = (_Float16*)(ws + off);              off += (size_t)HIDD * K3 * 2;
    _Float16* w4T = (_Float16*)(ws + off);              off += (size_t)IND * HIDD * 2;

    // 1) zero scatter target
    int aggN = NNODE * HIDD;
    zero_f32_kernel<<<(aggN + 255) / 256, 256, 0, stream>>>(agg, aggN);

    // 2) convert + transpose weights to f16 [N][Kpad]
    cvt_transpose_kernel<<<(HIDD * K1P + 255) / 256, 256, 0, stream>>>(w1, w1T, K1, HIDD, K1P);
    cvt_transpose_kernel<<<(HIDD * HIDD + 255) / 256, 256, 0, stream>>>(w2, w2T, HIDD, HIDD, HIDD);
    cvt_transpose_kernel<<<(HIDD * K3 + 255) / 256, 256, 0, stream>>>(w3, w3T, K3, HIDD, K3);
    cvt_transpose_kernel<<<(IND * HIDD + 255) / 256, 256, 0, stream>>>(w4, w4T, HIDD, IND, HIDD);

    // 3) edge message MLP + attention + scatter (5000 WGs)
    edge_msg_kernel<<<NEDGE / MT, 256, 0, stream>>>(h, ei, ea, w1T, b1, w2T, b2, wa, ba, agg);

    // 4) node update MLP + residual (tail-guarded)
    node_update_kernel<<<(NNODE + MT - 1) / MT, 256, 0, stream>>>(h, agg, w3T, b3, w4T, b4, out);
}
